// CausalSelfAttention_69028714381719
// MI455X (gfx1250) — matrix operand order repaired and checked by reading, not compiled, not run on hardware
//
#include <hip/hip_runtime.h>

// ---------------------------------------------------------------------------
// CDNA5 (gfx1250) bf16-WMMA causal self-attention (QKV proj + RoPE + flash
// attention + out proj). Wave32; all matmuls via v_wmma_f32_16x16x32_bf16.
// GEMM: 128x128 block tile, 32x64 wave tile, LDS double-buffered (1 barrier
// per k-step). Flash-attn tiles staged with global_load_async_to_lds_b128
// (ASYNCcnt path) + global prefetch of the next tile.
// Workspace requirement: ~96 MiB.
// ---------------------------------------------------------------------------

typedef __bf16 bf16;
typedef bf16  v16bf __attribute__((ext_vector_type(16)));
typedef float v8f   __attribute__((ext_vector_type(8)));

#define WMMA_BF16(a, b, c)                                                    \
  __builtin_amdgcn_wmma_f32_16x16x32_bf16(false, (a), false, (b), (short)0,   \
                                          (c), false, false)

// Async global->LDS copy (16B per lane), tracked by ASYNCcnt.
#define ASYNC_TO_LDS_B128(ldsOff32, gaddr64)                                  \
  asm volatile("global_load_async_to_lds_b128 %0, %1, off"                    \
               :: "v"(ldsOff32), "v"(gaddr64) : "memory")
#define WAIT_ASYNCCNT0() asm volatile("s_wait_asynccnt 0x0" ::: "memory")

static constexpr int BB = 2, TT = 2048, CC = 1024, HH = 16, DD = 64;

// ---- Fragment gathers (per CDNA5 ISA 7.12.2 layouts) ----------------------
// A 16x32 bf16, row-major src with leading dim ldk (elements).
__device__ __forceinline__ v16bf frag_a(const bf16* base, int ldk) {
  const int lane = threadIdx.x & 31;
  const int m = lane & 15;
  const int ko = (lane & 16) ? 4 : 0;  // dword offset (8 bf16)
  const unsigned* p = (const unsigned*)(base + m * ldk);
  union { v16bf v; unsigned u[8]; } f;
#pragma unroll
  for (int i = 0; i < 4; ++i) f.u[i] = p[i + ko];
#pragma unroll
  for (int i = 0; i < 4; ++i) f.u[4 + i] = p[8 + i + ko];
  return f.v;
}

// B 32x16 bf16 from a B^T ([n][k]) tile with leading dim ldk.
__device__ __forceinline__ v16bf frag_bT(const bf16* base, int ldk) {
  const int lane = threadIdx.x & 31;
  const int n = lane & 15;
  const int ko = (lane & 16) ? 4 : 0;
  const unsigned* p = (const unsigned*)(base + n * ldk);
  union { v16bf v; unsigned u[8]; } f;
#pragma unroll
  for (int i = 0; i < 4; ++i) f.u[i] = p[i + ko];
#pragma unroll
  for (int i = 0; i < 4; ++i) f.u[4 + i] = p[8 + i + ko];
  return f.v;
}

// ---- f32 -> bf16 conversion (8 elems/thread, b128 in / b128 out) ----------
__global__ void f32_to_bf16_k(const float* __restrict__ in,
                              bf16* __restrict__ out, int n) {
  int i = (blockIdx.x * blockDim.x + threadIdx.x) * 8;
  if (i >= n) return;
  float4 a = *(const float4*)(in + i);
  float4 b = *(const float4*)(in + i + 4);
  union { uint4 g; bf16 t[8]; } u;
  u.t[0] = (bf16)a.x; u.t[1] = (bf16)a.y; u.t[2] = (bf16)a.z; u.t[3] = (bf16)a.w;
  u.t[4] = (bf16)b.x; u.t[5] = (bf16)b.y; u.t[6] = (bf16)b.z; u.t[7] = (bf16)b.w;
  *(uint4*)(out + i) = u.g;
}

// ---- Tiled bf16 GEMM: C[f32, MxN] = A[MxK] * B[KxN] -----------------------
// 256 threads = 8 waves (4 M-waves x 2 N-waves); wave tile 32x64 = 8 WMMA per
// 32-wide k-step; LDS double-buffered via register staging (1 barrier/step).
// Requires M%128==0, N%128==0, K%32==0.
__global__ void __launch_bounds__(256)
gemm_bf16f32(const bf16* __restrict__ A, const bf16* __restrict__ Bm,
             float* __restrict__ Cg, int M, int N, int K) {
  constexpr int BM = 128, BN = 128, BK = 32;
  __shared__ __align__(16) bf16 As[2][BM * BK];
  __shared__ __align__(16) bf16 Bs[2][BN * BK];  // transposed [n][k]
  const int tid = threadIdx.x;
  const int wave = tid >> 5, lane = tid & 31;
  const int wm = wave & 3, wn = wave >> 2;
  const int hf = lane >> 4, ln = lane & 15;
  const int bm0 = blockIdx.x * BM, bn0 = blockIdx.y * BN;

  uint4 ra[2], rb[2];
  auto stage = [&](int k0) {  // global -> regs (4x b128 per thread)
#pragma unroll
    for (int v = 0; v < 2; ++v) {
      int p = tid + v * 256;
      int m = p >> 2, k = (p & 3) * 8;
      ra[v] = *(const uint4*)(A + (size_t)(bm0 + m) * K + k0 + k);
      int kk = p >> 4, n = (p & 15) * 8;
      rb[v] = *(const uint4*)(Bm + (size_t)(k0 + kk) * N + bn0 + n);
    }
  };
  auto commit = [&](int buf) {  // regs -> LDS (B transposed)
#pragma unroll
    for (int v = 0; v < 2; ++v) {
      int p = tid + v * 256;
      int m = p >> 2, k = (p & 3) * 8;
      *(uint4*)&As[buf][m * BK + k] = ra[v];
      int kk = p >> 4, n = (p & 15) * 8;
      union { uint4 g; bf16 t[8]; } u; u.g = rb[v];
#pragma unroll
      for (int j = 0; j < 8; ++j) Bs[buf][(n + j) * BK + kk] = u.t[j];
    }
  };

  v8f acc[2][4] = {};
  const int nsteps = K / BK;
  stage(0);
  commit(0);
  __syncthreads();
  for (int s = 0; s < nsteps; ++s) {
    const int cur = s & 1;
    if (s + 1 < nsteps) stage((s + 1) * BK);   // overlap HBM with WMMA
    v16bf af0 = frag_a(As[cur] + (wm * 32) * BK, BK);
    v16bf af1 = frag_a(As[cur] + (wm * 32 + 16) * BK, BK);
#pragma unroll
    for (int nt = 0; nt < 4; ++nt) {
      v16bf bfr = frag_bT(Bs[cur] + (wn * 64 + nt * 16) * BK, BK);
      acc[0][nt] = WMMA_BF16(af0, bfr, acc[0][nt]);
      acc[1][nt] = WMMA_BF16(af1, bfr, acc[1][nt]);
    }
    if (s + 1 < nsteps) commit(cur ^ 1);
    __syncthreads();
  }
#pragma unroll
  for (int mi = 0; mi < 2; ++mi)
#pragma unroll
    for (int nt = 0; nt < 4; ++nt) {
      int col = bn0 + wn * 64 + nt * 16 + ln;
#pragma unroll
      for (int r = 0; r < 8; ++r) {
        int row = bm0 + wm * 32 + mi * 16 + r + 8 * hf;
        Cg[(size_t)row * N + col] = acc[mi][nt][r];
      }
    }
}

// ---- RoPE + head split + layout (Q,K: [bh][t][64]; V^T: [bh][64][t]) ------
__global__ void rope_split_k(const float* __restrict__ qkv,
                             const float* __restrict__ fcos,
                             const float* __restrict__ fsin,
                             bf16* __restrict__ Qb, bf16* __restrict__ Kb,
                             bf16* __restrict__ Vt) {
  int tid = blockIdx.x * blockDim.x + threadIdx.x;  // B*T*H*32 = 2^21
  int p = tid & 31;
  int h = (tid >> 5) & 15;
  int t = (tid >> 9) & (TT - 1);
  int b = tid >> 20;
  size_t base = (size_t)(b * TT + t) * (3 * CC);
  int coff = h * DD + 2 * p;
  float c = fcos[t * 32 + p], s = fsin[t * 32 + p];
  float qe = qkv[base + coff], qo = qkv[base + coff + 1];
  float ke = qkv[base + CC + coff], ko = qkv[base + CC + coff + 1];
  float ve = qkv[base + 2 * CC + coff], vo = qkv[base + 2 * CC + coff + 1];
  int bh = b * HH + h;
  size_t qk = ((size_t)bh * TT + t) * DD + 2 * p;
  Qb[qk] = (bf16)(qe * c - qo * s);
  Qb[qk + 1] = (bf16)(qe * s + qo * c);
  Kb[qk] = (bf16)(ke * c - ko * s);
  Kb[qk + 1] = (bf16)(ke * s + ko * c);
  size_t vt0 = ((size_t)bh * DD + 2 * p) * TT + t;
  Vt[vt0] = (bf16)ve;
  Vt[vt0 + TT] = (bf16)vo;
}

// ---- Flash attention: grid (B*H, T/64), 128 threads (4 waves x 16 rows) ---
__global__ void __launch_bounds__(128)
flash_attn_k(const bf16* __restrict__ Qb, const bf16* __restrict__ Kb,
             const bf16* __restrict__ Vt, bf16* __restrict__ Ob) {
  __shared__ __align__(16) bf16 Ks[32 * 64];    // [key][d]  (= B^T for QK^T)
  __shared__ __align__(16) bf16 Vts[64 * 32];   // [d][key]  (= B^T for P*V)
  __shared__ __align__(16) bf16 Ps[4][16 * 32]; // per-wave P re-swizzle
  const int bh = blockIdx.x;
  const int qblk = blockIdx.y;
  const int tid = threadIdx.x;
  const int w = tid >> 5, lane = tid & 31;
  const int hf = lane >> 4, ln = lane & 15;
  const int q0 = qblk * 64 + w * 16;

  const bf16* qbase = Qb + ((size_t)bh * TT + q0) * DD;
  v16bf aq0 = frag_a(qbase, DD);        // D 0..31
  v16bf aq1 = frag_a(qbase + 32, DD);   // D 32..63

  v8f o[4] = {};
  float mrow[8], lrow[8];
#pragma unroll
  for (int r = 0; r < 8; ++r) { mrow[r] = -3.0e38f; lrow[r] = 0.f; }

  // Per-lane staging addresses (constant across kb except the tile offset).
  const int p0i = tid, p1i = tid + 128;
  const int krow0 = p0i >> 3, kc0 = (p0i & 7) * 8;
  const int krow1 = p1i >> 3, kc1 = (p1i & 7) * 8;
  const int vrow0 = p0i >> 2, vc0 = (p0i & 3) * 8;
  const int vrow1 = p1i >> 2, vc1 = (p1i & 3) * 8;
  const unsigned lk0 = (unsigned)(uintptr_t)&Ks[krow0 * 64 + kc0];
  const unsigned lk1 = (unsigned)(uintptr_t)&Ks[krow1 * 64 + kc1];
  const unsigned lv0 = (unsigned)(uintptr_t)&Vts[vrow0 * 32 + vc0];
  const unsigned lv1 = (unsigned)(uintptr_t)&Vts[vrow1 * 32 + vc1];

  const float scale = 0.125f;  // 1/sqrt(64)
  const int kb_end = qblk * 2 + 1;  // last 32-key block touching this q-tile
  for (int kb = 0; kb <= kb_end; ++kb) {
    {  // async global -> LDS staging of K tile [32][64] and V^T tile [64][32]
      const bf16* gk = Kb + ((size_t)bh * TT + kb * 32) * DD;
      const bf16* gv = Vt + (size_t)bh * DD * TT + kb * 32;
      ASYNC_TO_LDS_B128(lk0, (unsigned long long)(uintptr_t)(gk + krow0 * DD + kc0));
      ASYNC_TO_LDS_B128(lk1, (unsigned long long)(uintptr_t)(gk + krow1 * DD + kc1));
      ASYNC_TO_LDS_B128(lv0, (unsigned long long)(uintptr_t)(gv + (size_t)vrow0 * TT + vc0));
      ASYNC_TO_LDS_B128(lv1, (unsigned long long)(uintptr_t)(gv + (size_t)vrow1 * TT + vc1));
      if (kb < kb_end) {  // prefetch next tiles into cache
        __builtin_prefetch(gk + 32 * DD + krow0 * DD + kc0, 0, 3);
        __builtin_prefetch(gv + 32 + (size_t)vrow0 * TT + vc0, 0, 3);
      }
      WAIT_ASYNCCNT0();
    }
    __syncthreads();

    v8f s0 = {}, s1 = {};  // 16 queries x 32 keys (two 16-col groups)
    s0 = WMMA_BF16(aq0, frag_bT(Ks, 64), s0);
    s0 = WMMA_BF16(aq1, frag_bT(Ks + 32, 64), s0);
    s1 = WMMA_BF16(aq0, frag_bT(Ks + 16 * 64, 64), s1);
    s1 = WMMA_BF16(aq1, frag_bT(Ks + 16 * 64 + 32, 64), s1);

#pragma unroll
    for (int r = 0; r < 8; ++r) {
      int q = q0 + r + 8 * hf;               // this lane's row index
      int j0 = kb * 32 + ln, j1 = kb * 32 + 16 + ln;
      float v0 = (j0 <= q) ? s0[r] * scale : -3.0e38f;
      float v1 = (j1 <= q) ? s1[r] * scale : -3.0e38f;
      float mx = fmaxf(v0, v1);              // row reduce over 16-lane half
#pragma unroll
      for (int off = 1; off < 16; off <<= 1)
        mx = fmaxf(mx, __shfl_xor(mx, off, 32));
      float mnew = fmaxf(mrow[r], mx);
      float alpha = __expf(mrow[r] - mnew);
      mrow[r] = mnew;
      float p0 = __expf(v0 - mnew);
      float p1 = __expf(v1 - mnew);
      float ps = p0 + p1;
#pragma unroll
      for (int off = 1; off < 16; off <<= 1) ps += __shfl_xor(ps, off, 32);
      lrow[r] = lrow[r] * alpha + ps;
#pragma unroll
      for (int dg = 0; dg < 4; ++dg) o[dg][r] *= alpha;
      int m = r + 8 * hf;                    // C-layout -> A-layout via LDS
      Ps[w][m * 32 + ln] = (bf16)p0;
      Ps[w][m * 32 + 16 + ln] = (bf16)p1;
    }
    v16bf ap = frag_a(Ps[w], 32);            // P 16x32 bf16
#pragma unroll
    for (int dg = 0; dg < 4; ++dg)
      o[dg] = WMMA_BF16(ap, frag_bT(Vts + (dg * 16) * 32, 32), o[dg]);
    __syncthreads();
  }

  const int b = bh >> 4, h = bh & 15;
#pragma unroll
  for (int dg = 0; dg < 4; ++dg) {
#pragma unroll
    for (int r = 0; r < 8; ++r) {
      int q = q0 + r + 8 * hf;
      int c = h * DD + dg * 16 + ln;
      Ob[(size_t)(b * TT + q) * CC + c] = (bf16)(o[dg][r] / lrow[r]);
    }
  }
}

// ---------------------------------------------------------------------------
extern "C" void kernel_launch(void* const* d_in, const int* in_sizes, int n_in,
                              void* d_out, int out_size, void* d_ws,
                              size_t ws_size, hipStream_t stream) {
  (void)in_sizes; (void)n_in; (void)out_size; (void)ws_size;
  const float* x    = (const float*)d_in[0];
  const float* fcos = (const float*)d_in[1];
  const float* fsin = (const float*)d_in[2];
  const float* wqkv = (const float*)d_in[3];
  const float* wout = (const float*)d_in[4];
  float* out = (float*)d_out;

  char* ws = (char*)d_ws;
  size_t off = 0;
  bf16* xb    = (bf16*)(ws + off); off += (size_t)BB * TT * CC * 2;      // 8 MiB
  bf16* wqkvb = (bf16*)(ws + off); off += (size_t)CC * 3 * CC * 2;       // 6 MiB
  bf16* woutb = (bf16*)(ws + off); off += (size_t)CC * CC * 2;           // 2 MiB
  float* qkv  = (float*)(ws + off); off += (size_t)BB * TT * 3 * CC * 4; // 48 MiB
  bf16* Qb    = (bf16*)(ws + off); off += (size_t)BB * HH * TT * DD * 2; // 8 MiB
  bf16* Kb    = (bf16*)(ws + off); off += (size_t)BB * HH * TT * DD * 2; // 8 MiB
  bf16* Vt    = (bf16*)(ws + off); off += (size_t)BB * HH * DD * TT * 2; // 8 MiB
  bf16* Ab    = (bf16*)(ws + off); off += (size_t)BB * TT * CC * 2;      // 8 MiB

  f32_to_bf16_k<<<(BB * TT * CC) / (256 * 8), 256, 0, stream>>>(x, xb, BB * TT * CC);
  f32_to_bf16_k<<<(CC * 3 * CC) / (256 * 8), 256, 0, stream>>>(wqkv, wqkvb, CC * 3 * CC);
  f32_to_bf16_k<<<(CC * CC) / (256 * 8), 256, 0, stream>>>(wout, woutb, CC * CC);

  // qkv = x @ w_qkv    (M=4096, N=3072, K=1024)
  gemm_bf16f32<<<dim3((BB * TT) / 128, (3 * CC) / 128), 256, 0, stream>>>(
      xb, wqkvb, qkv, BB * TT, 3 * CC, CC);

  rope_split_k<<<(BB * TT * HH * 32) / 256, 256, 0, stream>>>(
      qkv, fcos, fsin, Qb, Kb, Vt);

  flash_attn_k<<<dim3(BB * HH, TT / 64), 128, 0, stream>>>(Qb, Kb, Vt, Ab);

  // out = attn @ w_out (M=4096, N=1024, K=1024)
  gemm_bf16f32<<<dim3((BB * TT) / 128, CC / 128), 256, 0, stream>>>(
      Ab, woutb, out, BB * TT, CC, CC);
}
